// GraphVAENF_2834678415604
// MI455X (gfx1250) — matrix operand order repaired and checked by reading, not compiled, not run on hardware
//
#include <hip/hip_runtime.h>
#include <hip/hip_bf16.h>

#define NNODES   49152
#define NEDGES   196608
#define BATCH    2048
#define HID      256
#define LATENT   128
#define FLOW_HID 256
#define NUMFLOWS 4
#define MAXATOMS 38
#define NNF      38
#define NEF      4

typedef __attribute__((ext_vector_type(16))) __bf16    v16bf;
typedef __attribute__((ext_vector_type(8)))  float     v8f;
typedef __attribute__((ext_vector_type(8)))  unsigned  v8u;
typedef __attribute__((ext_vector_type(4)))  unsigned  v4u;
typedef __attribute__((ext_vector_type(4)))  unsigned  uint32x4;
typedef __attribute__((ext_vector_type(8)))  int       int32x8;
typedef __attribute__((ext_vector_type(4)))  int       int32x4;

__device__ __forceinline__ unsigned bfbits(float f) {
  unsigned u = __builtin_bit_cast(unsigned, f);
  return (u + 0x7FFFu + ((u >> 16) & 1u)) >> 16;   // f32 -> bf16 RNE
}

// ---------------------------------------------------------------------------
// Conv-layer GEMM (M=NNODES, K=HID, N=HID), A pre-packed bf16.
// Whole Wt (256x256 bf16 = 128 KB) staged into LDS once per block via the
// Tensor Data Mover; each wave computes a 16-row strip across all 256 cols
// (16 WMMAs per A fragment). Output: either f32 (+relu) or bf16 (+relu).
// ---------------------------------------------------------------------------
__global__ void __launch_bounds__(256)
gemm_conv_wmma(const unsigned short* __restrict__ Abf,
               const unsigned short* __restrict__ Wt,
               const float* __restrict__ bias,
               float* __restrict__ OutF,            // nullable
               unsigned short* __restrict__ OutB)   // nullable
{
  __shared__ unsigned short sB[HID * HID];          // 128 KB

  // --- TDM: DMA the whole 256x256 bf16 weight matrix into LDS (wave 0) ---
  if (threadIdx.x < 32) {
    unsigned lds = (unsigned)(size_t)&sB[0];        // low 32 bits = LDS offset
    unsigned long long ga = (unsigned long long)(size_t)Wt;
    uint32x4 g0;
    g0[0] = 1u;                                     // count=1, user descriptor
    g0[1] = lds;                                    // lds_addr [63:32]
    g0[2] = (unsigned)ga;                           // global_addr lo
    g0[3] = (unsigned)((ga >> 32) & 0x01FFFFFFu) | 0x80000000u; // addr hi | type=2
    int32x8 g1;
    g1[0] = (int)(3u << 16);        // workgroup_mask=0, data_size=3 (8B units)
    g1[1] = (int)(64u << 16);       // tensor_dim0[15:0]=64 (512B row / 8B)
    g1[2] = (int)(256u << 16);      // tensor_dim0 hi=0 | tensor_dim1 lo=256
    g1[3] = (int)(64u << 16);       // tensor_dim1 hi=0 | tile_dim0=64
    g1[4] = (int)256;               // tile_dim1=256, tile_dim2=0
    g1[5] = (int)64;                // tensor_dim0_stride lo = 64
    g1[6] = 0;                      // stride0 hi | dim1_stride lo
    g1[7] = 0;
    int32x4 gz4 = {0, 0, 0, 0};
    int32x8 gz8 = {0, 0, 0, 0, 0, 0, 0, 0};
    __builtin_amdgcn_tensor_load_to_lds(g0, g1, gz4, gz4, gz8, 0);
    __builtin_amdgcn_s_wait_tensorcnt(0);
  }
  __syncthreads();

  int mt   = blockIdx.x * 8 + (threadIdx.x >> 5);   // grid sized so mt < M/16
  int lane = threadIdx.x & 31;
  int m    = (mt << 4) + (lane & 15);
  int kb   = (lane >> 4) << 3;                      // 0 or 8
  int krow = (lane & 15) + ((lane >> 4) << 4);      // B row within k-tile

  const size_t rowA = (size_t)m * HID;
  v8f acc[16] = {};

  for (int k0 = 0; k0 < HID; k0 += 32) {
    v4u lo = *(const v4u*)(Abf + rowA + k0 + kb);        // K = kb..kb+7
    v4u hi = *(const v4u*)(Abf + rowA + k0 + kb + 16);   // K = kb+16..kb+23
    v8u au;
    au[0] = lo[0]; au[1] = lo[1]; au[2] = lo[2]; au[3] = lo[3];
    au[4] = hi[0]; au[5] = hi[1]; au[6] = hi[2]; au[7] = hi[3];
    v16bf a = __builtin_bit_cast(v16bf, au);
    const unsigned short* brow = sB + (size_t)(k0 + kb) * HID + (lane & 15);
#pragma unroll
    for (int nt = 0; nt < 16; ++nt) {
      v16bf b;
#pragma unroll
      for (int s = 0; s < 16; ++s)
        b[s] = __builtin_bit_cast(__bf16, brow[(size_t)(s + (s & 8)) * HID + (nt << 4)]);
      acc[nt] = __builtin_amdgcn_wmma_f32_16x16x32_bf16(
          false, a, false, b, (short)0, acc[nt], false, false);
    }
  }

  int r0 = (mt << 4) + ((lane >> 4) << 3);
  int nb = lane & 15;
#pragma unroll
  for (int nt = 0; nt < 16; ++nt) {
    int n = (nt << 4) + nb;
    float bv = bias[n];
#pragma unroll
    for (int r = 0; r < 8; ++r) {
      float v = fmaxf(acc[nt][r] + bv, 0.f);
      if (OutF) OutF[(size_t)(r0 + r) * HID + n] = v;
      else      OutB[(size_t)(r0 + r) * HID + n] = (unsigned short)bfbits(v);
    }
  }
}

// ---------------------------------------------------------------------------
// Generic GEMM (f32 A, converted inline): Out[M,Nout] = act(A@Wt + bias).
// Each wave: one 16x64 strip (4 WMMAs per A fragment). Npad % 64 == 0.
// ---------------------------------------------------------------------------
__global__ void gemm_wmma_f32a(const float* __restrict__ A,
                               const unsigned short* __restrict__ Wt,
                               const float* __restrict__ bias,
                               float* __restrict__ Out,
                               int M, int K, int Npad, int Nout,
                               int relu, int totWaves)
{
  int wave = blockIdx.x * (blockDim.x >> 5) + (threadIdx.x >> 5);
  if (wave >= totWaves) return;                 // wave-uniform
  int ngroups = Npad >> 6;
  int ng = wave % ngroups;
  int mt = wave / ngroups;
  int lane = threadIdx.x & 31;

  int m    = (mt << 4) + (lane & 15);
  int kb   = (lane >> 4) << 3;
  int krow = (lane & 15) + ((lane >> 4) << 4);
  const size_t rowA = (size_t)m * (size_t)K;

  v8f acc[4] = {};
  for (int k0 = 0; k0 < K; k0 += 32) {
    const float* ap = A + rowA + k0 + kb;
    float4 c0 = *(const float4*)(ap);
    float4 c1 = *(const float4*)(ap + 4);
    float4 c2 = *(const float4*)(ap + 16);
    float4 c3 = *(const float4*)(ap + 20);
    __builtin_prefetch(ap + 32, 0, 0);
    v8u au;
    au[0] = bfbits(c0.x) | (bfbits(c0.y) << 16);
    au[1] = bfbits(c0.z) | (bfbits(c0.w) << 16);
    au[2] = bfbits(c1.x) | (bfbits(c1.y) << 16);
    au[3] = bfbits(c1.z) | (bfbits(c1.w) << 16);
    au[4] = bfbits(c2.x) | (bfbits(c2.y) << 16);
    au[5] = bfbits(c2.z) | (bfbits(c2.w) << 16);
    au[6] = bfbits(c3.x) | (bfbits(c3.y) << 16);
    au[7] = bfbits(c3.z) | (bfbits(c3.w) << 16);
    v16bf a = __builtin_bit_cast(v16bf, au);

    const unsigned short* wp = Wt + (size_t)(k0 + kb) * (size_t)Npad + (ng << 6) + (lane & 15);
#pragma unroll
    for (int nt = 0; nt < 4; ++nt) {
      v16bf b;
#pragma unroll
      for (int s = 0; s < 16; ++s)
        b[s] = __builtin_bit_cast(__bf16, wp[(size_t)(s + (s & 8)) * (size_t)Npad + (nt << 4)]);
      acc[nt] = __builtin_amdgcn_wmma_f32_16x16x32_bf16(
          false, a, false, b, (short)0, acc[nt], false, false);
    }
  }

  int r0 = (mt << 4) + ((lane >> 4) << 3);
#pragma unroll
  for (int nt = 0; nt < 4; ++nt) {
    int n = (ng << 6) + (nt << 4) + (lane & 15);
    if (n < Nout) {
      float bv = bias ? bias[n] : 0.f;
      for (int r = 0; r < 8; ++r) {
        float v = acc[nt][r] + bv;
        if (relu) v = fmaxf(v, 0.f);
        Out[(size_t)(r0 + r) * (size_t)Nout + n] = v;
      }
    }
  }
}

// ---------------------------------------------------------------------------
// Weight prep: W[out,in] f32 -> Wt[in,Npad] bf16 (zero-padded cols; MADE masks)
// ---------------------------------------------------------------------------
__global__ void prep_wt(const float* __restrict__ W, unsigned short* __restrict__ Wt,
                        int out_dim, int in_dim, int Npad, int mode)
{
  int idx = blockIdx.x * blockDim.x + threadIdx.x;
  if (idx >= in_dim * Npad) return;
  int o = idx % Npad, k = idx / Npad;
  float w = 0.f;
  if (o < out_dim) {
    w = W[(size_t)o * in_dim + k];
    if (mode == 1) {        // mask1[h=o][i=k]: keep if (o%127)+1 >= k+1
      if (((o % (LATENT - 1)) + 1) < (k + 1)) w = 0.f;
    } else if (mode == 2) { // mask2[o][h=k]: keep if (o%128)+1 > (k%127)+1
      if (((o % LATENT) + 1) <= ((k % (LATENT - 1)) + 1)) w = 0.f;
    }
  }
  Wt[idx] = (unsigned short)bfbits(w);
}

// ---------------------------------------------------------------------------
__global__ void zero_f32(float* __restrict__ p, long n) {
  long i = (long)blockIdx.x * blockDim.x + threadIdx.x;
  if (i < n) p[i] = 0.f;
}

__global__ void copy_f32(float* __restrict__ dst, const float* __restrict__ src, long n) {
  long i = (long)blockIdx.x * blockDim.x + threadIdx.x;
  if (i < n) dst[i] = src[i];
}

// hin_bf = bf16(h + aggr), packed pairs
__global__ void combine_bf16(const float* __restrict__ h, const float* __restrict__ aggr,
                             unsigned* __restrict__ outb, long npairs)
{
  long i = (long)blockIdx.x * blockDim.x + threadIdx.x;
  if (i >= npairs) return;
  float2 a = *(const float2*)(h + i * 2);
  float2 b = *(const float2*)(aggr + i * 2);
  outb[i] = bfbits(a.x + b.x) | (bfbits(a.y + b.y) << 16);
}

__global__ void embed_nodes(const int* __restrict__ x, const float* __restrict__ emb,
                            float* __restrict__ h)
{
  int idx = blockIdx.x * blockDim.x + threadIdx.x;
  if (idx >= NNODES * (HID / 4)) return;
  int n = idx >> 6, c4 = (idx & 63) << 2;
  *(float4*)(h + (size_t)n * HID + c4) =
      *(const float4*)(emb + (size_t)x[n] * HID + c4);
}

__global__ void msg_aggr(const int* __restrict__ src, const int* __restrict__ dst,
                         const int* __restrict__ eattr,
                         const float* __restrict__ h, const float* __restrict__ edge_emb,
                         float* __restrict__ aggr)
{
  int idx = blockIdx.x * blockDim.x + threadIdx.x;
  if (idx >= NEDGES * (HID / 4)) return;
  int e = idx >> 6, c4 = (idx & 63) << 2;
  int s = src[e], d = dst[e], a = eattr[e];
  float4 hv = *(const float4*)(h + (size_t)s * HID + c4);
  float4 ev = *(const float4*)(edge_emb + (size_t)a * HID + c4);
  float4 m;
  m.x = fmaxf(hv.x + ev.x, 0.f); m.y = fmaxf(hv.y + ev.y, 0.f);
  m.z = fmaxf(hv.z + ev.z, 0.f); m.w = fmaxf(hv.w + ev.w, 0.f);
  float* p = aggr + (size_t)d * HID + c4;
  __hip_atomic_fetch_add(p + 0, m.x, __ATOMIC_RELAXED, __HIP_MEMORY_SCOPE_AGENT);
  __hip_atomic_fetch_add(p + 1, m.y, __ATOMIC_RELAXED, __HIP_MEMORY_SCOPE_AGENT);
  __hip_atomic_fetch_add(p + 2, m.z, __ATOMIC_RELAXED, __HIP_MEMORY_SCOPE_AGENT);
  __hip_atomic_fetch_add(p + 3, m.w, __ATOMIC_RELAXED, __HIP_MEMORY_SCOPE_AGENT);
}

__global__ void pool_nodes(const float* __restrict__ h, const int* __restrict__ batch,
                           float* __restrict__ hg)
{
  int idx = blockIdx.x * blockDim.x + threadIdx.x;
  if (idx >= NNODES * (HID / 4)) return;
  int n = idx >> 6, c4 = (idx & 63) << 2;
  int b = batch[n];
  float4 v = *(const float4*)(h + (size_t)n * HID + c4);
  float* p = hg + (size_t)b * HID + c4;
  __hip_atomic_fetch_add(p + 0, v.x, __ATOMIC_RELAXED, __HIP_MEMORY_SCOPE_AGENT);
  __hip_atomic_fetch_add(p + 1, v.y, __ATOMIC_RELAXED, __HIP_MEMORY_SCOPE_AGENT);
  __hip_atomic_fetch_add(p + 2, v.z, __ATOMIC_RELAXED, __HIP_MEMORY_SCOPE_AGENT);
  __hip_atomic_fetch_add(p + 3, v.w, __ATOMIC_RELAXED, __HIP_MEMORY_SCOPE_AGENT);
}

__global__ void z0_kernel(const float* __restrict__ mu, const float* __restrict__ lv,
                          const float* __restrict__ eps,
                          float* __restrict__ z0_out, float* __restrict__ z)
{
  int i = blockIdx.x * blockDim.x + threadIdx.x;
  if (i >= BATCH * LATENT) return;
  float v = mu[i] + eps[i] * expf(0.5f * lv[i]);
  z0_out[i] = v;
  z[i] = v;
}

__global__ void flow_update(const float* __restrict__ fout,
                            float* __restrict__ z, float* __restrict__ sld)
{
  int b = blockIdx.x, j = threadIdx.x;     // blockDim.x == LATENT
  float m = fout[(size_t)b * (2 * LATENT) + j];
  float s = fout[(size_t)b * (2 * LATENT) + LATENT + j];
  float gate = 1.f / (1.f + expf(-s));
  float zj = z[(size_t)b * LATENT + j];
  z[(size_t)b * LATENT + j] = gate * zj + (1.f - gate) * m;
  __shared__ float red[LATENT];
  red[j] = logf(gate + 1e-8f);
  __syncthreads();
  for (int off = LATENT / 2; off > 0; off >>= 1) {
    if (j < off) red[j] += red[j + off];
    __syncthreads();
  }
  if (j == 0) sld[b] += red[0];
}

__global__ void symmetrize(float* __restrict__ el)
{
  const int PAIRS = MAXATOMS * (MAXATOMS + 1) / 2;   // 741
  int idx = blockIdx.x * blockDim.x + threadIdx.x;
  if (idx >= BATCH * PAIRS * NEF) return;
  int c = idx % NEF;
  int t = idx / NEF;
  int p = t % PAIRS, b = t / PAIRS;
  int i = 0, rem = p;
  while (rem >= (MAXATOMS - i)) { rem -= (MAXATOMS - i); ++i; }
  int j = i + rem;
  size_t base = (size_t)b * (MAXATOMS * MAXATOMS * NEF);
  size_t a1 = base + ((size_t)i * MAXATOMS + j) * NEF + c;
  size_t a2 = base + ((size_t)j * MAXATOMS + i) * NEF + c;
  float v = 0.5f * (el[a1] + el[a2]);
  el[a1] = v; el[a2] = v;
}

// ---------------------------------------------------------------------------
extern "C" void kernel_launch(void* const* d_in, const int* in_sizes, int n_in,
                              void* d_out, int out_size, void* d_ws, size_t ws_size,
                              hipStream_t stream)
{
  (void)in_sizes; (void)n_in; (void)out_size; (void)ws_size;
  const int*   x        = (const int*)  d_in[0];
  const int*   eidx     = (const int*)  d_in[1];
  const int*   eattr    = (const int*)  d_in[2];
  const int*   batch    = (const int*)  d_in[3];
  const float* eps      = (const float*)d_in[4];
  const float* node_emb = (const float*)d_in[5];
  const float* edge_emb = (const float*)d_in[6];
  const float* conv_w1  = (const float*)d_in[7];
  const float* conv_b1  = (const float*)d_in[8];
  const float* conv_w2  = (const float*)d_in[9];
  const float* conv_b2  = (const float*)d_in[10];
  const float* mu_w     = (const float*)d_in[11];
  const float* mu_b     = (const float*)d_in[12];
  const float* lv_w     = (const float*)d_in[13];
  const float* lv_b     = (const float*)d_in[14];
  const float* flow_w1  = (const float*)d_in[15];
  const float* flow_b1  = (const float*)d_in[16];
  const float* flow_w2  = (const float*)d_in[17];
  const float* flow_b2  = (const float*)d_in[18];
  const float* dn_w1    = (const float*)d_in[19];
  const float* dn_b1    = (const float*)d_in[20];
  const float* dn_w2    = (const float*)d_in[21];
  const float* dn_b2    = (const float*)d_in[22];
  const float* de_w1    = (const float*)d_in[23];
  const float* de_b1    = (const float*)d_in[24];
  const float* de_w2    = (const float*)d_in[25];
  const float* de_b2    = (const float*)d_in[26];

  const int* esrc = eidx;
  const int* edst = eidx + NEDGES;

  // ---- output regions ----
  float* out = (float*)d_out;
  size_t off_edge = (size_t)BATCH * MAXATOMS * NNF;
  size_t off_mu   = off_edge + (size_t)BATCH * MAXATOMS * MAXATOMS * NEF;
  size_t off_lv   = off_mu + (size_t)BATCH * LATENT;
  size_t off_z0   = off_lv + (size_t)BATCH * LATENT;
  size_t off_zK   = off_z0 + (size_t)BATCH * LATENT;
  size_t off_sld  = off_zK + (size_t)BATCH * LATENT;
  float* out_node = out;
  float* out_edge = out + off_edge;
  float* out_mu   = out + off_mu;
  float* out_lv   = out + off_lv;
  float* out_z0   = out + off_z0;
  float* out_zK   = out + off_zK;
  float* out_sld  = out + off_sld;

  // ---- workspace bump allocator ----
  char* wsb = (char*)d_ws;
  size_t off = 0;
  auto alloc = [&](size_t bytes) -> void* {
    void* p = wsb + off;
    off = (off + bytes + 255) & ~(size_t)255;
    return p;
  };
  float* h      = (float*)alloc((size_t)NNODES * HID * 4);
  float* aggr   = (float*)alloc((size_t)NNODES * HID * 4);
  unsigned short* hin_bf = (unsigned short*)alloc((size_t)NNODES * HID * 2);
  unsigned short* t2b    = (unsigned short*)alloc((size_t)NNODES * HID * 2);
  float* hgraph = (float*)alloc((size_t)BATCH * HID * 4);
  float* z      = (float*)alloc((size_t)BATCH * LATENT * 4);
  float* h1     = (float*)alloc((size_t)BATCH * FLOW_HID * 4);
  float* fout   = (float*)alloc((size_t)BATCH * 2 * LATENT * 4);
  float* he     = (float*)alloc((size_t)BATCH * 512 * 4);
  unsigned short* conv_w1t = (unsigned short*)alloc((size_t)4 * HID * HID * 2);
  unsigned short* conv_w2t = (unsigned short*)alloc((size_t)4 * HID * HID * 2);
  unsigned short* mu_wt    = (unsigned short*)alloc((size_t)HID * LATENT * 2);
  unsigned short* lv_wt    = (unsigned short*)alloc((size_t)HID * LATENT * 2);
  unsigned short* flow_w1t = (unsigned short*)alloc((size_t)4 * LATENT * FLOW_HID * 2);
  unsigned short* flow_w2t = (unsigned short*)alloc((size_t)4 * FLOW_HID * 2 * LATENT * 2);
  unsigned short* dn_w1t   = (unsigned short*)alloc((size_t)LATENT * 256 * 2);
  unsigned short* dn_w2t   = (unsigned short*)alloc((size_t)256 * 1472 * 2);
  unsigned short* de_w1t   = (unsigned short*)alloc((size_t)LATENT * 512 * 2);
  unsigned short* de_w2t   = (unsigned short*)alloc((size_t)512 * 5824 * 2);

  auto cdiv = [](long a, long b) { return (int)((a + b - 1) / b); };

  auto prep = [&](const float* W, unsigned short* Wt, int od, int id, int Npad, int mode) {
    prep_wt<<<cdiv((long)id * Npad, 256), 256, 0, stream>>>(W, Wt, od, id, Npad, mode);
  };
  auto gemm = [&](const float* A, const unsigned short* Wt, const float* bias,
                  float* Out, int M, int K, int Npad, int Nout, int relu) {
    int waves = (M / 16) * (Npad / 64);
    gemm_wmma_f32a<<<cdiv(waves, 8), 256, 0, stream>>>(A, Wt, bias, Out,
                                                       M, K, Npad, Nout, relu, waves);
  };

  // ---- weight prep ----
  for (int k = 0; k < 4; ++k) {
    prep(conv_w1 + (size_t)k * HID * HID, conv_w1t + (size_t)k * HID * HID, HID, HID, HID, 0);
    prep(conv_w2 + (size_t)k * HID * HID, conv_w2t + (size_t)k * HID * HID, HID, HID, HID, 0);
    prep(flow_w1 + (size_t)k * FLOW_HID * LATENT,
         flow_w1t + (size_t)k * LATENT * FLOW_HID, FLOW_HID, LATENT, FLOW_HID, 1);
    prep(flow_w2 + (size_t)k * 2 * LATENT * FLOW_HID,
         flow_w2t + (size_t)k * FLOW_HID * 2 * LATENT, 2 * LATENT, FLOW_HID, 2 * LATENT, 2);
  }
  prep(mu_w, mu_wt, LATENT, HID, LATENT, 0);
  prep(lv_w, lv_wt, LATENT, HID, LATENT, 0);
  prep(dn_w1, dn_w1t, 256, LATENT, 256, 0);
  prep(dn_w2, dn_w2t, MAXATOMS * NNF, 256, 1472, 0);
  prep(de_w1, de_w1t, 512, LATENT, 512, 0);
  prep(de_w2, de_w2t, MAXATOMS * MAXATOMS * NEF, 512, 5824, 0);

  // ---- GNN ----
  embed_nodes<<<cdiv((long)NNODES * 64, 256), 256, 0, stream>>>(x, node_emb, h);
  for (int k = 0; k < 4; ++k) {
    zero_f32<<<cdiv((long)NNODES * HID, 256), 256, 0, stream>>>(aggr, (long)NNODES * HID);
    msg_aggr<<<cdiv((long)NEDGES * 64, 256), 256, 0, stream>>>(esrc, edst, eattr, h, edge_emb, aggr);
    combine_bf16<<<cdiv((long)NNODES * HID / 2, 256), 256, 0, stream>>>(
        h, aggr, (unsigned*)hin_bf, (long)NNODES * HID / 2);
    // t2b = bf16(relu(hin @ W1^T + b1))
    gemm_conv_wmma<<<NNODES / 16 / 8, 256, 0, stream>>>(
        hin_bf, conv_w1t + (size_t)k * HID * HID, conv_b1 + (size_t)k * HID, nullptr, t2b);
    // h = relu(t2b @ W2^T + b2)
    gemm_conv_wmma<<<NNODES / 16 / 8, 256, 0, stream>>>(
        t2b, conv_w2t + (size_t)k * HID * HID, conv_b2 + (size_t)k * HID, h, nullptr);
  }

  // ---- pool + VAE head ----
  zero_f32<<<cdiv((long)BATCH * HID, 256), 256, 0, stream>>>(hgraph, (long)BATCH * HID);
  pool_nodes<<<cdiv((long)NNODES * 64, 256), 256, 0, stream>>>(h, batch, hgraph);
  gemm(hgraph, mu_wt, mu_b, out_mu, BATCH, HID, LATENT, LATENT, 0);
  gemm(hgraph, lv_wt, lv_b, out_lv, BATCH, HID, LATENT, LATENT, 0);
  z0_kernel<<<cdiv((long)BATCH * LATENT, 256), 256, 0, stream>>>(out_mu, out_lv, eps, out_z0, z);
  zero_f32<<<cdiv((long)BATCH, 256), 256, 0, stream>>>(out_sld, (long)BATCH);

  // ---- MADE flows ----
  for (int k = 0; k < NUMFLOWS; ++k) {
    gemm(z, flow_w1t + (size_t)k * LATENT * FLOW_HID,
         flow_b1 + (size_t)k * FLOW_HID, h1, BATCH, LATENT, FLOW_HID, FLOW_HID, 1);
    gemm(h1, flow_w2t + (size_t)k * FLOW_HID * 2 * LATENT,
         flow_b2 + (size_t)k * 2 * LATENT, fout, BATCH, FLOW_HID, 2 * LATENT, 2 * LATENT, 0);
    flow_update<<<BATCH, LATENT, 0, stream>>>(fout, z, out_sld);
  }
  copy_f32<<<cdiv((long)BATCH * LATENT, 256), 256, 0, stream>>>(out_zK, z, (long)BATCH * LATENT);

  // ---- decoders ----
  gemm(z, dn_w1t, dn_b1, h1, BATCH, LATENT, 256, 256, 1);
  gemm(h1, dn_w2t, dn_b2, out_node, BATCH, 256, 1472, MAXATOMS * NNF, 0);
  gemm(z, de_w1t, de_b1, he, BATCH, LATENT, 512, 512, 1);
  gemm(he, de_w2t, de_b2, out_edge, BATCH, 512, 5824, MAXATOMS * MAXATOMS * NEF, 0);
  symmetrize<<<cdiv((long)BATCH * 741 * NEF, 256), 256, 0, stream>>>(out_edge);
}
